// SlotAttentionTranslRotScaleEquiv_22428319220326
// MI455X (gfx1250) — compile-verified, matrix operand order checked by reading, not run on hardware
//
#include <hip/hip_runtime.h>
#include <hip/hip_bf16.h>

// ---------------------------------------------------------------------------
// SlotAttention (transl/rot/scale equivariant) for gfx1250 / MI455X, wave32.
// - k_rel GEMM folded into per-slot vectors (qg = Wge q)
// - one fused attention pass per iteration (WMMA logits + WMMA weighted-V)
// - TDM tensor_load_to_lds stages the k-slab into LDS in the attention pass
// - GRU + MLP slot update fully on WMMA
// ---------------------------------------------------------------------------

#define BB 8
#define SS 16
#define NN 1024
#define DD 128
#define MLPH 256
#define ITERS 3

typedef __attribute__((ext_vector_type(16))) _Float16 v16h;
typedef __attribute__((ext_vector_type(8)))  float    v8f;
typedef __attribute__((ext_vector_type(4)))  unsigned v4u;
typedef __attribute__((ext_vector_type(8)))  int      v8i;
typedef __attribute__((ext_vector_type(4)))  int      v4i;

#if __has_builtin(__builtin_amdgcn_tensor_load_to_lds) && \
    __has_builtin(__builtin_amdgcn_s_wait_tensorcnt)
#define HAVE_TDM 1
#endif

__device__ __forceinline__ v8f wmma_f16(v16h a, v16h b, v8f c) {
  // D = A(16x32 f16) * B(32x16 f16) + C(16x16 f32)
  return __builtin_amdgcn_wmma_f32_16x16x32_f16(
      false, a, false, b, (short)0, c, false, false);
}

// Load a 16x32 f16 fragment (A layout per CDNA5 ISA 7.12.2).
// p is [row][k] row-major with `stride` halves per row; n0 = first row.
// Serves B operands too when the array is stored [n][k]-major.
__device__ __forceinline__ v16h ld_tile16(const _Float16* p, int n0, int stride,
                                          int kb, int lane) {
  union { v16h v; unsigned u[8]; } a;
  const _Float16* base =
      p + (size_t)(n0 + (lane & 15)) * stride + kb + ((lane < 16) ? 0 : 8);
#pragma unroll
  for (int i = 0; i < 8; ++i)
    a.u[i] = *(const unsigned*)(base + ((i < 4) ? 0 : 16) + 2 * (i & 3));
  return a.v;
}

// ---------------------------------------------------------------------------
// Weight conversion to f16. Projection weights transposed to [out][in] so B
// fragment K-pairs are contiguous. Wge stays row-major ([n][k] for q@WgeT).
// ---------------------------------------------------------------------------
__global__ void k_convert(const float* __restrict__ Wq, const float* __restrict__ Wk,
                          const float* __restrict__ Wv, const float* __restrict__ Wge,
                          const float* __restrict__ Wir, const float* __restrict__ Wiz,
                          const float* __restrict__ Win, const float* __restrict__ Whr,
                          const float* __restrict__ Whz, const float* __restrict__ Whn,
                          const float* __restrict__ W1, const float* __restrict__ W2,
                          _Float16* __restrict__ WqT, _Float16* __restrict__ WkT,
                          _Float16* __restrict__ WvT, _Float16* __restrict__ Wge16,
                          _Float16* __restrict__ WirT, _Float16* __restrict__ WizT,
                          _Float16* __restrict__ WinT, _Float16* __restrict__ WhrT,
                          _Float16* __restrict__ WhzT, _Float16* __restrict__ WhnT,
                          _Float16* __restrict__ W1T, _Float16* __restrict__ W2T) {
  const int i = blockIdx.x * 256 + threadIdx.x;  // 0 .. 32767
  if (i < DD * DD) {
    const int r = i >> 7, c = i & 127;
    WqT[c * DD + r] = (_Float16)Wq[i];
    WkT[c * DD + r] = (_Float16)Wk[i];
    WvT[c * DD + r] = (_Float16)Wv[i];
    Wge16[i]        = (_Float16)Wge[i];
    WirT[c * DD + r] = (_Float16)Wir[i];
    WizT[c * DD + r] = (_Float16)Wiz[i];
    WinT[c * DD + r] = (_Float16)Win[i];
    WhrT[c * DD + r] = (_Float16)Whr[i];
    WhzT[c * DD + r] = (_Float16)Whz[i];
    WhnT[c * DD + r] = (_Float16)Whn[i];
  }
  if (i < DD * MLPH) {
    { const int r = i / MLPH, c = i % MLPH; W1T[c * DD + r] = (_Float16)W1[i]; }
    { const int r = i / DD, c = i % DD; W2T[c * MLPH + r] = (_Float16)W2[i]; }
  }
}

// ---------------------------------------------------------------------------
__global__ void k_init(const float* __restrict__ slots, float* __restrict__ slots_cur,
                       float* __restrict__ pos, float* __restrict__ scl,
                       float* __restrict__ rotm) {
  const int bs = blockIdx.x, t = threadIdx.x;
  slots_cur[bs * DD + t] = slots[bs * 136 + t];
  if (t < 2) { pos[bs * 2 + t] = 0.f; scl[bs * 2 + t] = 0.1f; }
  if (t < 4) rotm[bs * 4 + t] = slots[bs * 136 + 132 + t];
}

// Grid moments (grid identical across batch): G1[2], G2[00,01,11].
__global__ void k_gmom(const float* __restrict__ inp, float* __restrict__ gmom) {
  __shared__ float red[256];
  const int t = threadIdx.x;
  float m[5] = {0.f, 0.f, 0.f, 0.f, 0.f};
  for (int n = t; n < NN; n += 256) {
    const float g0 = inp[(size_t)n * 130 + 128];
    const float g1 = inp[(size_t)n * 130 + 129];
    m[0] += g0; m[1] += g1; m[2] += g0 * g0; m[3] += g0 * g1; m[4] += g1 * g1;
  }
  for (int q = 0; q < 5; ++q) {
    red[t] = m[q]; __syncthreads();
    for (int off = 128; off; off >>= 1) {
      if (t < off) red[t] += red[t + off];
      __syncthreads();
    }
    if (t == 0) gmom[q] = red[0];
    __syncthreads();
  }
}

// LayerNorm feats (rows = B*N), write f16.
__global__ void k_ln_feats(const float* __restrict__ inp, const float* __restrict__ sc,
                           const float* __restrict__ bi, _Float16* __restrict__ out) {
  __shared__ float red[128];
  const int row = blockIdx.x, t = threadIdx.x;
  const float v = inp[(size_t)row * 130 + t];
  red[t] = v; __syncthreads();
  for (int off = 64; off; off >>= 1) { if (t < off) red[t] += red[t + off]; __syncthreads(); }
  const float mean = red[0] * (1.f / DD); __syncthreads();
  red[t] = v * v; __syncthreads();
  for (int off = 64; off; off >>= 1) { if (t < off) red[t] += red[t + off]; __syncthreads(); }
  const float var = red[0] * (1.f / DD) - mean * mean;
  out[(size_t)row * DD + t] =
      (_Float16)((v - mean) * rsqrtf(var + 1e-6f) * sc[t] + bi[t]);
}

// ---------------------------------------------------------------------------
// k = lnfeats @ Wk (-> k16 [b][n][d]) ; v = lnfeats @ Wv (-> vT16 [b][d][n]).
// ---------------------------------------------------------------------------
__global__ void __launch_bounds__(256)
k_gemm_kv(const _Float16* __restrict__ lnf, const _Float16* __restrict__ WkT,
          const _Float16* __restrict__ WvT, _Float16* __restrict__ k16,
          _Float16* __restrict__ vT16) {
  const int b = blockIdx.x >> 3, slab = blockIdx.x & 7;
  const int wave = threadIdx.x >> 5, lane = threadIdx.x & 31;
  const int keyb = slab * 128 + wave * 16;
  const _Float16* A = lnf + (size_t)b * NN * DD;
  __builtin_prefetch(A + (size_t)(keyb + 16) * DD, 0, 0);
  const v16h a0 = ld_tile16(A, keyb, DD, 0, lane);
  const v16h a1 = ld_tile16(A, keyb, DD, 32, lane);
  const v16h a2 = ld_tile16(A, keyb, DD, 64, lane);
  const v16h a3 = ld_tile16(A, keyb, DD, 96, lane);
  const int rowOff = (lane < 16) ? 0 : 8, col = lane & 15;
  for (int m = 0; m < 2; ++m) {
    const _Float16* W = m ? WvT : WkT;
    for (int jt = 0; jt < 8; ++jt) {
      v8f c = {};
      c = wmma_f16(a0, ld_tile16(W, jt * 16, DD, 0, lane), c);
      c = wmma_f16(a1, ld_tile16(W, jt * 16, DD, 32, lane), c);
      c = wmma_f16(a2, ld_tile16(W, jt * 16, DD, 64, lane), c);
      c = wmma_f16(a3, ld_tile16(W, jt * 16, DD, 96, lane), c);
      if (m == 0) {
#pragma unroll
        for (int r = 0; r < 8; ++r)
          k16[(size_t)b * NN * DD + (size_t)(keyb + rowOff + r) * DD + jt * 16 + col] =
              (_Float16)c[r];
      } else {
#pragma unroll
        for (int r = 0; r < 8; ++r)
          vT16[(size_t)b * DD * NN + (size_t)(jt * 16 + col) * NN + keyb + rowOff + r] =
              (_Float16)c[r];
      }
    }
  }
}

// ---------------------------------------------------------------------------
// Per-iteration slot prep (one block, rows = B*S = 128):
//  q  = LN(slots) @ Wq / sqrt(128);  qg = q @ WgeT
//  wg2[s] = Wgp qg[s];  const[s] = bgp.qg[s] + bge.q[s]
// ---------------------------------------------------------------------------
__global__ void __launch_bounds__(256)
k_slot_prep(const float* __restrict__ slots_cur, const float* __restrict__ lnqs,
            const float* __restrict__ lnqb, const _Float16* __restrict__ WqT,
            const _Float16* __restrict__ Wge16, const float* __restrict__ Wgp,
            const float* __restrict__ bgp, const float* __restrict__ bge,
            _Float16* __restrict__ qg16, float* __restrict__ wg2,
            float* __restrict__ constL) {
  __shared__ _Float16 bufA[128 * 128];
  __shared__ _Float16 bufB[128 * 128];
  __shared__ float cq[128], cb[128], w2[256];
  const int t = threadIdx.x, wave = t >> 5, lane = t & 31;
  if (t < 128) { cq[t] = 0.f; cb[t] = 0.f; w2[2 * t] = 0.f; w2[2 * t + 1] = 0.f; }
  if (t < 128) {
    const float* x = slots_cur + t * DD;
    float s = 0.f, s2 = 0.f;
    for (int j = 0; j < DD; ++j) { const float v = x[j]; s += v; s2 += v * v; }
    const float m = s * (1.f / DD);
    const float inv = rsqrtf(s2 * (1.f / DD) - m * m + 1e-6f);
    for (int j = 0; j < DD; ++j)
      bufA[t * DD + j] = (_Float16)((x[j] - m) * inv * lnqs[j] + lnqb[j]);
  }
  __syncthreads();
  const int row = wave * 16, rowOff = (lane < 16) ? 0 : 8, col = lane & 15;
  {  // q phase
    const v16h a0 = ld_tile16(bufA, row, DD, 0, lane);
    const v16h a1 = ld_tile16(bufA, row, DD, 32, lane);
    const v16h a2 = ld_tile16(bufA, row, DD, 64, lane);
    const v16h a3 = ld_tile16(bufA, row, DD, 96, lane);
    for (int jt = 0; jt < 8; ++jt) {
      v8f c = {};
      c = wmma_f16(a0, ld_tile16(WqT, jt * 16, DD, 0, lane), c);
      c = wmma_f16(a1, ld_tile16(WqT, jt * 16, DD, 32, lane), c);
      c = wmma_f16(a2, ld_tile16(WqT, jt * 16, DD, 64, lane), c);
      c = wmma_f16(a3, ld_tile16(WqT, jt * 16, DD, 96, lane), c);
#pragma unroll
      for (int r = 0; r < 8; ++r) {
        const int rr = row + rowOff + r, jj = jt * 16 + col;
        const float qv = c[r] * 0.08838834764831845f;  // 1/sqrt(128)
        bufB[rr * DD + jj] = (_Float16)qv;
        atomicAdd(&cq[rr], qv * bge[jj]);
      }
    }
  }
  __syncthreads();
  {  // qg phase
    const v16h a0 = ld_tile16(bufB, row, DD, 0, lane);
    const v16h a1 = ld_tile16(bufB, row, DD, 32, lane);
    const v16h a2 = ld_tile16(bufB, row, DD, 64, lane);
    const v16h a3 = ld_tile16(bufB, row, DD, 96, lane);
    for (int jt = 0; jt < 8; ++jt) {
      v8f c = {};
      c = wmma_f16(a0, ld_tile16(Wge16, jt * 16, DD, 0, lane), c);
      c = wmma_f16(a1, ld_tile16(Wge16, jt * 16, DD, 32, lane), c);
      c = wmma_f16(a2, ld_tile16(Wge16, jt * 16, DD, 64, lane), c);
      c = wmma_f16(a3, ld_tile16(Wge16, jt * 16, DD, 96, lane), c);
#pragma unroll
      for (int r = 0; r < 8; ++r) {
        const int rr = row + rowOff + r, jj = jt * 16 + col;
        const float g = c[r];
        qg16[rr * DD + jj] = (_Float16)g;
        atomicAdd(&cb[rr], g * bgp[jj]);
        atomicAdd(&w2[rr * 2 + 0], g * Wgp[jj]);
        atomicAdd(&w2[rr * 2 + 1], g * Wgp[DD + jj]);
      }
    }
  }
  __syncthreads();
  if (t < 128) {
    constL[t] = cq[t] + cb[t];
    wg2[2 * t] = w2[2 * t];
    wg2[2 * t + 1] = w2[2 * t + 1];
  }
}

__global__ void k_clear(float* __restrict__ acc) {
  acc[blockIdx.x * 256 + threadIdx.x] = 0.f;
}

// ---------------------------------------------------------------------------
// Attention pass: 32 blocks (B x N/256), 256 thr.
// Phase 0: TDM tensor_load_to_lds of the 256x128 f16 k-slab (64 KB)
// Phase 1: WMMA logits0[key,s] = k . qg  -> LDS
// Phase 2: positional term + softmax over S per key; w -> LDS f16; moments
// Phase 3: WMMA sum_n w[s,n] v[n,d] -> global f32 atomics into acc
// acc row layout per (b,s): [sum_w, swg0, swg1, swg00, swg01, swg11, swv[128]]
// ---------------------------------------------------------------------------
__global__ void __launch_bounds__(256)
k_attn(const _Float16* __restrict__ k16, const _Float16* __restrict__ qg16,
       const _Float16* __restrict__ vT16, const float* __restrict__ inp,
       const float* __restrict__ pos, const float* __restrict__ scl,
       const float* __restrict__ rotm, const float* __restrict__ wg2,
       const float* __restrict__ constL, float* __restrict__ acc) {
  const int b = blockIdx.x >> 2, nblk = blockIdx.x & 3;
  const int kbase = nblk * 256;
  __shared__ _Float16 kbuf[256 * DD];  // 64 KB staged k-slab
  __shared__ float lg[256 * 16];
  __shared__ _Float16 wl[16 * 256];
  __shared__ float part[16 * 6];
  __shared__ float sp[16 * 12];
  const int t = threadIdx.x, wave = t >> 5, lane = t & 31;
  if (t < 96) part[t] = 0.f;
  if (t < 16) {
    const int bs = b * SS + t;
    sp[t * 12 + 0] = pos[bs * 2 + 0]; sp[t * 12 + 1] = pos[bs * 2 + 1];
    sp[t * 12 + 2] = rotm[bs * 4 + 0]; sp[t * 12 + 3] = rotm[bs * 4 + 1];
    sp[t * 12 + 4] = rotm[bs * 4 + 2]; sp[t * 12 + 5] = rotm[bs * 4 + 3];
    sp[t * 12 + 6] = 1.f / scl[bs * 2 + 0]; sp[t * 12 + 7] = 1.f / scl[bs * 2 + 1];
    sp[t * 12 + 8] = wg2[bs * 2 + 0]; sp[t * 12 + 9] = wg2[bs * 2 + 1];
    sp[t * 12 + 10] = constL[bs]; sp[t * 12 + 11] = 0.f;
  }
#ifdef HAVE_TDM
  if (wave == 0) {
    // D# : count=1, type=2 ("image"); 1-D tile of 32768 f16 elements.
    const _Float16* gsrc = k16 + (size_t)b * NN * DD + (size_t)kbase * DD;
    const unsigned long long ga = (unsigned long long)(size_t)gsrc;
    const unsigned lds_off = (unsigned)(size_t)(void*)&kbuf[0];
    v4u g0 = {1u, lds_off, (unsigned)ga,
              (unsigned)((ga >> 32) & 0x01FFFFFFu) | (2u << 30)};
    // group1: data_size=2B (bits17:16); tensor_dim0=32768 (bits79:48);
    // tensor_dim1=1 (bits111:80); tile_dim0=32768 (bits127:112); rest 0.
    v8i g1 = {(int)(1u << 16), (int)0x80000000u, (int)0x00010000u,
              (int)0x80000000u, 0, 0, 0, 0};
    v4i g2 = {0, 0, 0, 0};
    v4i g3 = {0, 0, 0, 0};
#if defined(__clang_major__) && __clang_major__ >= 23
    v8i g4 = {0, 0, 0, 0, 0, 0, 0, 0};
    __builtin_amdgcn_tensor_load_to_lds(g0, g1, g2, g3, g4, 0);
#else
    __builtin_amdgcn_tensor_load_to_lds(g0, g1, g2, g3, 0);
#endif
    __builtin_amdgcn_s_wait_tensorcnt(0);
  }
#else
  for (int i = t; i < 256 * DD; i += 256)
    kbuf[i] = k16[(size_t)b * NN * DD + (size_t)kbase * DD + i];
#endif
  __syncthreads();
  const int rowOff = (lane < 16) ? 0 : 8, col = lane & 15;
  const _Float16* Qg = qg16 + b * SS * DD;
#pragma unroll
  for (int h = 0; h < 2; ++h) {  // each wave: two 16-key tiles
    const int keyl = wave * 32 + h * 16;  // local row in kbuf
    v8f c = {};
#pragma unroll
    for (int ks = 0; ks < 4; ++ks) {
      const v16h a = ld_tile16(kbuf, keyl, DD, ks * 32, lane);
      const v16h bm = ld_tile16(Qg, 0, DD, ks * 32, lane);
      c = wmma_f16(a, bm, c);
    }
#pragma unroll
    for (int r = 0; r < 8; ++r)
      lg[(keyl + rowOff + r) * 16 + col] = c[r];
  }
  __syncthreads();
  {  // per-key softmax over the 16 slots
    const int key = kbase + t;
    const float g0 = inp[((size_t)(b * NN + key)) * 130 + 128];
    const float g1 = inp[((size_t)(b * NN + key)) * 130 + 129];
    float lv[16];
    float mx = -3.0e38f;
#pragma unroll
    for (int s = 0; s < 16; ++s) {
      const float* q = &sp[s * 12];
      const float r0 = g0 - q[0], r1 = g1 - q[1];
      const float x0 = (q[2] * r0 + q[3] * r1) * q[6];
      const float x1 = (q[4] * r0 + q[5] * r1) * q[7];
      const float l = lg[t * 16 + s] + x0 * q[8] + x1 * q[9] + q[10];
      lv[s] = l; mx = fmaxf(mx, l);
    }
    float se = 0.f;
#pragma unroll
    for (int s = 0; s < 16; ++s) { lv[s] = __expf(lv[s] - mx); se += lv[s]; }
    const float inv = 1.f / se;
#pragma unroll
    for (int s = 0; s < 16; ++s) {
      const float w = lv[s] * inv;
      wl[s * 256 + t] = (_Float16)w;
      atomicAdd(&part[s * 6 + 0], w);
      atomicAdd(&part[s * 6 + 1], w * g0);
      atomicAdd(&part[s * 6 + 2], w * g1);
      atomicAdd(&part[s * 6 + 3], w * g0 * g0);
      atomicAdd(&part[s * 6 + 4], w * g0 * g1);
      atomicAdd(&part[s * 6 + 5], w * g1 * g1);
    }
  }
  __syncthreads();
  if (t < 96) atomicAdd(&acc[(b * SS + t / 6) * 136 + (t % 6)], part[t]);
  {  // weighted value sum: A = w (16 x 256), B = vT (keys x d cols)
    const int dbase = wave * 16;
    const _Float16* Vb = vT16 + (size_t)b * DD * NN;
    v8f c = {};
#pragma unroll
    for (int ks = 0; ks < 8; ++ks) {
      const v16h a = ld_tile16(wl, 0, 256, ks * 32, lane);
      const v16h bm = ld_tile16(Vb, dbase, NN, kbase + ks * 32, lane);
      c = wmma_f16(a, bm, c);
    }
#pragma unroll
    for (int r = 0; r < 8; ++r) {
      const int slot = rowOff + r, d = dbase + col;
      atomicAdd(&acc[(b * SS + slot) * 136 + 6 + d], c[r]);
    }
  }
}

// ---------------------------------------------------------------------------
// Per-slot update: normalize, positions, covariance eigenframe, scales.
// Second moments reconstructed from slot moments + fixed grid moments.
// ---------------------------------------------------------------------------
__global__ void k_slot_update(const float* __restrict__ acc, const float* __restrict__ gmom,
                              float* __restrict__ updates, float* __restrict__ pos,
                              float* __restrict__ scl, float* __restrict__ rotm) {
  const int bs = blockIdx.x, t = threadIdx.x;
  const float* a = acc + bs * 136;
  const float norm = a[0] + 1e-8f;
  updates[bs * DD + t] = a[6 + t] / norm;
  if (t == 0) {
    const float inv = 1.f / norm;
    const float sA = a[0] * inv;
    const float p0 = a[1] * inv, p1 = a[2] * inv;
    const float A200 = a[3] * inv, A201 = a[4] * inv, A211 = a[5] * inv;
    const float C00 = A200 - 2.f * p0 * p0 + sA * p0 * p0;
    const float C01 = A201 - 2.f * p0 * p1 + sA * p0 * p1;
    const float C11 = A211 - 2.f * p1 * p1 + sA * p1 * p1;
    const float cov00 = C11, cov01 = C01, cov11 = C00;  // xs = swapped comps
    const float pm = sqrtf(4.f * cov01 * cov01 + (cov00 - cov11) * (cov00 - cov11) + 1e-16f);
    const float eig1 = 0.5f * (cov00 + cov11 + pm);
    const float eig2 = 0.5f * (cov00 + cov11 - pm);
    const float div = cov01 + 1e-16f;
    float v1x, v1y, v2x, v2y;
    if (fabsf(div) < 1e-6f) { v1x = 1.f; v1y = 0.f; v2x = 0.f; v2y = 1.f; }
    else { v1x = (eig1 - cov11) / div; v1y = 1.f; v2x = (eig2 - cov11) / div; v2y = 1.f; }
    const float n1 = rsqrtf(v1x * v1x + v1y * v1y); v1x *= n1; v1y *= n1;
    const float n2 = rsqrtf(v2x * v2x + v2y * v2y); v2x *= n2; v2y *= n2;
    if (v1x < 0.707f && v1x > -0.707f) {
      const float tx = v1x, ty = v1y; v1x = v2x; v1y = v2y; v2x = tx; v2y = ty;
    }
    if (v1x < 0.f) { v1x = -v1x; v1y = -v1y; }
    rotm[bs * 4 + 0] = v1x; rotm[bs * 4 + 1] = v1y;
    rotm[bs * 4 + 2] = v2x; rotm[bs * 4 + 3] = v2y;
    pos[bs * 2 + 0] = p0; pos[bs * 2 + 1] = p1;
    const float G10 = gmom[0], G11 = gmom[1];
    const float G200 = gmom[2], G201 = gmom[3], G211 = gmom[4];
    const float Nf = (float)NN;
    const float D00 = G200 - 2.f * p0 * G10 + Nf * p0 * p0;
    const float D01 = G201 - p0 * G11 - p1 * G10 + Nf * p0 * p1;
    const float D11 = G211 - 2.f * p1 * G11 + Nf * p1 * p1;
    const float M00 = C00 + 1e-8f * D00;
    const float M01 = C01 + 1e-8f * D01;
    const float M11 = C11 + 1e-8f * D11;
    float s0 = sqrtf(fmaxf(v1x * v1x * M00 + 2.f * v1x * v1y * M01 + v1y * v1y * M11, 0.f));
    float s1 = sqrtf(fmaxf(v2x * v2x * M00 + 2.f * v2x * v2y * M01 + v2y * v2y * M11, 0.f));
    scl[bs * 2 + 0] = fminf(fmaxf(s0, 0.001f), 2.f);
    scl[bs * 2 + 1] = fminf(fmaxf(s1, 0.001f), 2.f);
  }
}

// ---------------------------------------------------------------------------
// GRU + residual MLP on the 128 slot rows — fully WMMA.
//  r = sig(u@Wir + h@Whr + bhr); z = sig(u@Wiz + h@Whz + bhz)
//  n = tanh(u@Win + r*(h@Whn + bhn)); ns = (1-z)n + z h
//  out = ns + relu(LN(ns)@W1 + b1)@W2 + b2
// hid activations round-trip through f16 scratch; ns through f32 scratch.
// ---------------------------------------------------------------------------
__global__ void __launch_bounds__(256)
k_gru_mlp(const float* __restrict__ updates, float* __restrict__ slots_cur,
          const _Float16* __restrict__ WirT, const _Float16* __restrict__ WizT,
          const _Float16* __restrict__ WinT, const _Float16* __restrict__ WhrT,
          const _Float16* __restrict__ WhzT, const _Float16* __restrict__ WhnT,
          const float* __restrict__ bhr, const float* __restrict__ bhz,
          const float* __restrict__ bhn, const float* __restrict__ lnms,
          const float* __restrict__ lnmb, const _Float16* __restrict__ W1T,
          const float* __restrict__ b1, const _Float16* __restrict__ W2T,
          const float* __restrict__ b2, float* __restrict__ nsbuf,
          _Float16* __restrict__ hid16) {
  __shared__ _Float16 bufU[128 * 128];
  __shared__ _Float16 bufH[128 * 128];
  const int t = threadIdx.x, wave = t >> 5, lane = t & 31;
  for (int i = t; i < 128 * 128; i += 256) {
    bufU[i] = (_Float16)updates[i];
    bufH[i] = (_Float16)slots_cur[i];
  }
  __syncthreads();
  const int row = wave * 16, rowOff = (lane < 16) ? 0 : 8, col = lane & 15;
  {  // gate phase
    const v16h u0 = ld_tile16(bufU, row, DD, 0, lane);
    const v16h u1 = ld_tile16(bufU, row, DD, 32, lane);
    const v16h u2 = ld_tile16(bufU, row, DD, 64, lane);
    const v16h u3 = ld_tile16(bufU, row, DD, 96, lane);
    const v16h h0 = ld_tile16(bufH, row, DD, 0, lane);
    const v16h h1 = ld_tile16(bufH, row, DD, 32, lane);
    const v16h h2 = ld_tile16(bufH, row, DD, 64, lane);
    const v16h h3 = ld_tile16(bufH, row, DD, 96, lane);
    for (int jt = 0; jt < 8; ++jt) {
      v8f cr = {}, cz = {}, ca = {}, ch = {};
      cr = wmma_f16(u0, ld_tile16(WirT, jt * 16, DD, 0, lane), cr);
      cr = wmma_f16(u1, ld_tile16(WirT, jt * 16, DD, 32, lane), cr);
      cr = wmma_f16(u2, ld_tile16(WirT, jt * 16, DD, 64, lane), cr);
      cr = wmma_f16(u3, ld_tile16(WirT, jt * 16, DD, 96, lane), cr);
      cr = wmma_f16(h0, ld_tile16(WhrT, jt * 16, DD, 0, lane), cr);
      cr = wmma_f16(h1, ld_tile16(WhrT, jt * 16, DD, 32, lane), cr);
      cr = wmma_f16(h2, ld_tile16(WhrT, jt * 16, DD, 64, lane), cr);
      cr = wmma_f16(h3, ld_tile16(WhrT, jt * 16, DD, 96, lane), cr);
      cz = wmma_f16(u0, ld_tile16(WizT, jt * 16, DD, 0, lane), cz);
      cz = wmma_f16(u1, ld_tile16(WizT, jt * 16, DD, 32, lane), cz);
      cz = wmma_f16(u2, ld_tile16(WizT, jt * 16, DD, 64, lane), cz);
      cz = wmma_f16(u3, ld_tile16(WizT, jt * 16, DD, 96, lane), cz);
      cz = wmma_f16(h0, ld_tile16(WhzT, jt * 16, DD, 0, lane), cz);
      cz = wmma_f16(h1, ld_tile16(WhzT, jt * 16, DD, 32, lane), cz);
      cz = wmma_f16(h2, ld_tile16(WhzT, jt * 16, DD, 64, lane), cz);
      cz = wmma_f16(h3, ld_tile16(WhzT, jt * 16, DD, 96, lane), cz);
      ca = wmma_f16(u0, ld_tile16(WinT, jt * 16, DD, 0, lane), ca);
      ca = wmma_f16(u1, ld_tile16(WinT, jt * 16, DD, 32, lane), ca);
      ca = wmma_f16(u2, ld_tile16(WinT, jt * 16, DD, 64, lane), ca);
      ca = wmma_f16(u3, ld_tile16(WinT, jt * 16, DD, 96, lane), ca);
      ch = wmma_f16(h0, ld_tile16(WhnT, jt * 16, DD, 0, lane), ch);
      ch = wmma_f16(h1, ld_tile16(WhnT, jt * 16, DD, 32, lane), ch);
      ch = wmma_f16(h2, ld_tile16(WhnT, jt * 16, DD, 64, lane), ch);
      ch = wmma_f16(h3, ld_tile16(WhnT, jt * 16, DD, 96, lane), ch);
#pragma unroll
      for (int r = 0; r < 8; ++r) {
        const int rr = row + rowOff + r, jj = jt * 16 + col;
        const float rg = 1.f / (1.f + __expf(-(cr[r] + bhr[jj])));
        const float zg = 1.f / (1.f + __expf(-(cz[r] + bhz[jj])));
        const float ng = tanhf(ca[r] + rg * (ch[r] + bhn[jj]));
        const float hv = (float)bufH[rr * DD + jj];
        nsbuf[rr * DD + jj] = (1.f - zg) * ng + zg * hv;
      }
    }
  }
  __syncthreads();
  if (t < 128) {  // LayerNorm of ns rows -> bufU (f16)
    const float* x = nsbuf + t * DD;
    float s = 0.f, s2 = 0.f;
    for (int j = 0; j < DD; ++j) { const float v = x[j]; s += v; s2 += v * v; }
    const float m = s * (1.f / DD);
    const float inv = rsqrtf(s2 * (1.f / DD) - m * m + 1e-6f);
    for (int j = 0; j < DD; ++j)
      bufU[t * DD + j] = (_Float16)((x[j] - m) * inv * lnms[j] + lnmb[j]);
  }
  __syncthreads();
  {  // MLP layer 1: hid = relu(LN @ W1 + b1), N = 256
    const v16h a0 = ld_tile16(bufU, row, DD, 0, lane);
    const v16h a1 = ld_tile16(bufU, row, DD, 32, lane);
    const v16h a2 = ld_tile16(bufU, row, DD, 64, lane);
    const v16h a3 = ld_tile16(bufU, row, DD, 96, lane);
    for (int jt = 0; jt < 16; ++jt) {
      v8f c = {};
      c = wmma_f16(a0, ld_tile16(W1T, jt * 16, DD, 0, lane), c);
      c = wmma_f16(a1, ld_tile16(W1T, jt * 16, DD, 32, lane), c);
      c = wmma_f16(a2, ld_tile16(W1T, jt * 16, DD, 64, lane), c);
      c = wmma_f16(a3, ld_tile16(W1T, jt * 16, DD, 96, lane), c);
#pragma unroll
      for (int r = 0; r < 8; ++r) {
        const int rr = row + rowOff + r, jj = jt * 16 + col;
        hid16[rr * MLPH + jj] = (_Float16)fmaxf(c[r] + b1[jj], 0.f);
      }
    }
  }
  __syncthreads();
  {  // MLP layer 2: out = ns + hid @ W2 + b2, K = 256
    for (int jt = 0; jt < 8; ++jt) {
      v8f c = {};
      for (int ks = 0; ks < 8; ++ks)
        c = wmma_f16(ld_tile16(hid16, row, MLPH, ks * 32, lane),
                     ld_tile16(W2T, jt * 16, MLPH, ks * 32, lane), c);
#pragma unroll
      for (int r = 0; r < 8; ++r) {
        const int rr = row + rowOff + r, jj = jt * 16 + col;
        slots_cur[rr * DD + jj] = nsbuf[rr * DD + jj] + c[r] + b2[jj];
      }
    }
  }
}

__global__ void k_final(const float* __restrict__ slots_cur, const float* __restrict__ pos,
                        const float* __restrict__ scl, const float* __restrict__ rotm,
                        float* __restrict__ out) {
  const int bs = blockIdx.x, t = threadIdx.x;
  float v;
  if (t < 128) v = slots_cur[bs * DD + t];
  else if (t < 130) v = pos[bs * 2 + (t - 128)];
  else if (t < 132) v = scl[bs * 2 + (t - 130)];
  else if (t < 136) v = rotm[bs * 4 + (t - 132)];
  else return;
  out[bs * 136 + t] = v;
}

// ---------------------------------------------------------------------------
extern "C" void kernel_launch(void* const* d_in, const int* in_sizes, int n_in,
                              void* d_out, int out_size, void* d_ws, size_t ws_size,
                              hipStream_t stream) {
  const float* slots  = (const float*)d_in[0];
  const float* inputs = (const float*)d_in[1];
  const float* ln_in_s = (const float*)d_in[2];
  const float* ln_in_b = (const float*)d_in[3];
  const float* Wq = (const float*)d_in[4];
  const float* Wk = (const float*)d_in[5];
  const float* Wv = (const float*)d_in[6];
  const float* Wgp = (const float*)d_in[7];
  const float* bgp = (const float*)d_in[8];
  const float* Wge = (const float*)d_in[9];
  const float* bge = (const float*)d_in[10];
  const float* ln_q_s = (const float*)d_in[11];
  const float* ln_q_b = (const float*)d_in[12];
  const float* Wir = (const float*)d_in[13];
  const float* Wiz = (const float*)d_in[14];
  const float* Win = (const float*)d_in[15];
  const float* Whr = (const float*)d_in[16];
  const float* Whz = (const float*)d_in[17];
  const float* Whn = (const float*)d_in[18];
  const float* bhr = (const float*)d_in[19];
  const float* bhz = (const float*)d_in[20];
  const float* bhn = (const float*)d_in[21];
  const float* ln_m_s = (const float*)d_in[22];
  const float* ln_m_b = (const float*)d_in[23];
  const float* W1 = (const float*)d_in[24];
  const float* b1 = (const float*)d_in[25];
  const float* W2 = (const float*)d_in[26];
  const float* b2 = (const float*)d_in[27];
  (void)in_sizes; (void)n_in; (void)out_size; (void)ws_size;

  char* ws = (char*)d_ws;
  auto take = [&](size_t bytes) {
    char* p = ws;
    ws += (bytes + 255) & ~(size_t)255;
    return p;
  };
  _Float16* lnf   = (_Float16*)take((size_t)BB * NN * DD * 2);
  _Float16* k16   = (_Float16*)take((size_t)BB * NN * DD * 2);
  _Float16* vT16  = (_Float16*)take((size_t)BB * NN * DD * 2);
  _Float16* WqT   = (_Float16*)take(DD * DD * 2);
  _Float16* WkT   = (_Float16*)take(DD * DD * 2);
  _Float16* WvT   = (_Float16*)take(DD * DD * 2);
  _Float16* Wge16 = (_Float16*)take(DD * DD * 2);
  _Float16* WirT  = (_Float16*)take(DD * DD * 2);
  _Float16* WizT  = (_Float16*)take(DD * DD * 2);
  _Float16* WinT  = (_Float16*)take(DD * DD * 2);
  _Float16* WhrT  = (_Float16*)take(DD * DD * 2);
  _Float16* WhzT  = (_Float16*)take(DD * DD * 2);
  _Float16* WhnT  = (_Float16*)take(DD * DD * 2);
  _Float16* W1T   = (_Float16*)take(DD * MLPH * 2);
  _Float16* W2T   = (_Float16*)take(DD * MLPH * 2);
  _Float16* qg16  = (_Float16*)take(BB * SS * DD * 2);
  _Float16* hid16 = (_Float16*)take(BB * SS * MLPH * 2);
  float* slots_cur = (float*)take(BB * SS * DD * 4);
  float* updates   = (float*)take(BB * SS * DD * 4);
  float* nsbuf     = (float*)take(BB * SS * DD * 4);
  float* pos  = (float*)take(BB * SS * 2 * 4);
  float* scl  = (float*)take(BB * SS * 2 * 4);
  float* rotm = (float*)take(BB * SS * 4 * 4);
  float* wg2  = (float*)take(BB * SS * 2 * 4);
  float* constL = (float*)take(BB * SS * 4);
  float* acc  = (float*)take(BB * SS * 136 * 4);
  float* gmom = (float*)take(8 * 4);

  k_convert<<<128, 256, 0, stream>>>(Wq, Wk, Wv, Wge, Wir, Wiz, Win, Whr, Whz, Whn,
                                     W1, W2, WqT, WkT, WvT, Wge16, WirT, WizT, WinT,
                                     WhrT, WhzT, WhnT, W1T, W2T);
  k_init<<<BB * SS, DD, 0, stream>>>(slots, slots_cur, pos, scl, rotm);
  k_gmom<<<1, 256, 0, stream>>>(inputs, gmom);
  k_ln_feats<<<BB * NN, DD, 0, stream>>>(inputs, ln_in_s, ln_in_b, lnf);
  k_gemm_kv<<<64, 256, 0, stream>>>(lnf, WkT, WvT, k16, vT16);

  for (int it = 0; it <= ITERS; ++it) {
    k_clear<<<(BB * SS * 136) / 256, 256, 0, stream>>>(acc);
    k_slot_prep<<<1, 256, 0, stream>>>(slots_cur, ln_q_s, ln_q_b, WqT, Wge16,
                                       Wgp, bgp, bge, qg16, wg2, constL);
    k_attn<<<BB * (NN / 256), 256, 0, stream>>>(k16, qg16, vT16, inputs, pos, scl,
                                                rotm, wg2, constL, acc);
    k_slot_update<<<BB * SS, DD, 0, stream>>>(acc, gmom, updates, pos, scl, rotm);
    if (it < ITERS)
      k_gru_mlp<<<1, 256, 0, stream>>>(updates, slots_cur, WirT, WizT, WinT, WhrT,
                                       WhzT, WhnT, bhr, bhz, bhn, ln_m_s, ln_m_b,
                                       W1T, b1, W2T, b2, nsbuf, hid16);
  }
  k_final<<<BB * SS, 160, 0, stream>>>(slots_cur, pos, scl, rotm, (float*)d_out);
}
